// QuantizedLinearLayer_82944408420940
// MI455X (gfx1250) — matrix operand order repaired and checked by reading, not compiled, not run on hardware
//
#include <hip/hip_runtime.h>
#include <hip/hip_fp16.h>

typedef __attribute__((ext_vector_type(16))) _Float16 v16h;
typedef __attribute__((ext_vector_type(8)))  float    v8f;
typedef __attribute__((ext_vector_type(4)))  unsigned int u32x4;
typedef __attribute__((ext_vector_type(4)))  int i32x4;
typedef __attribute__((ext_vector_type(8)))  int i32x8;

#define K_DIM 4096
#define N_DIM 4096
#define BM 128
#define BN 256
#define BK 64
#define LDSP 72   // padded LDS row stride in halfs (64 data + 8 pad = 144 B)

union V16 { v16h v; uint4 q[2]; };
union PK4 { _Float16 h[4]; uint2 u; };
union PK8 { _Float16 h[8]; uint4 u; };

// ---------------------------------------------------------------------------
// Pre-pass: int32-held int8 weights -> f16 (exact; scale deferred to epilogue)
// ---------------------------------------------------------------------------
__global__ __launch_bounds__(256)
void w_to_f16_kernel(const int* __restrict__ q, _Float16* __restrict__ wh, int n) {
    int idx = (blockIdx.x * 256 + threadIdx.x) * 8;
    if (idx >= n) return;
    int4 a = *(const int4*)(q + idx);
    int4 b = *(const int4*)(q + idx + 4);
    PK8 pk;
    pk.h[0] = (_Float16)a.x; pk.h[1] = (_Float16)a.y;
    pk.h[2] = (_Float16)a.z; pk.h[3] = (_Float16)a.w;
    pk.h[4] = (_Float16)b.x; pk.h[5] = (_Float16)b.y;
    pk.h[6] = (_Float16)b.z; pk.h[7] = (_Float16)b.w;
    *(uint4*)(wh + idx) = pk.u;
}

// ---------------------------------------------------------------------------
// TDM: DMA one 256(n) x 64(k) f16 weight tile from global into LDS,
// inserting 16 B of padding per 128 B row (-> 144 B LDS row stride = LDSP).
// D# layout per CDNA5 ISA ch.8 (group0: count/lds/global/type; group1: dims).
// ---------------------------------------------------------------------------
__device__ __forceinline__ void tdm_load_b_tile(const _Float16* gsrc, unsigned lds_off) {
    unsigned long long ga = (unsigned long long)(const void*)gsrc;
    u32x4 g0;
    g0[0] = 1u;                                   // count=1 (valid user descriptor)
    g0[1] = lds_off;                              // lds_addr (bytes)
    g0[2] = (unsigned)(ga & 0xffffffffu);         // global_addr[31:0]
    g0[3] = (unsigned)((ga >> 32) & 0x01ffffffu)  // global_addr[56:32]
          | (2u << 30);                           // type = 2 ("image")
    i32x8 g1;
    g1[0] = (1 << 16)                             // data_size = 2 bytes
          | (1 << 20)                             // pad_enable
          | (4 << 22)                             // pad_interval: 32 DWORDs (one 64-half row)
          | (3 << 25);                            // pad_amount: 4 DWORDs (16 B)
    g1[1] = (int)((unsigned)K_DIM << 16);         // tensor_dim0 = K (low 16 -> bits[63:48])
    g1[2] = (int)((unsigned)N_DIM << 16);         // tensor_dim0 hi=0 | tensor_dim1 low16
    g1[3] = (int)(64u << 16);                     // tensor_dim1 hi=0 | tile_dim0 = 64
    g1[4] = BN;                                   // tile_dim1 = 256 | tile_dim2 = 0
    g1[5] = K_DIM;                                // tensor_dim0_stride[31:0] = K
    g1[6] = 0;                                    // stride hi | dim1_stride lo
    g1[7] = 0;
    i32x4 z4 = {0, 0, 0, 0};
#if __clang_major__ >= 23
    i32x8 z8 = {0, 0, 0, 0, 0, 0, 0, 0};
    __builtin_amdgcn_tensor_load_to_lds(g0, g1, z4, z4, z8, 0);
#else
    __builtin_amdgcn_tensor_load_to_lds(g0, g1, z4, z4, 0);
#endif
}

// ---------------------------------------------------------------------------
// Main GEMM: y = x @ Wf16^T * scale[n] + bias[n], f32 accumulate via WMMA.
// Block tile 128x256, 8 waves in a 2(M) x 4(N) grid, 64x64 per wave
// (16 WMMAs from 4 A + 4 B fragments per k-step -> 32 WMMAs per K-tile).
// Pipeline: TDM loads B tile kt+1; A tile kt+1 prefetched to regs; one
// barrier per K-tile with double-buffered LDS.
// ---------------------------------------------------------------------------
__global__ __launch_bounds__(256)
void qlinear_wmma_kernel(const float* __restrict__ X,
                         const _Float16* __restrict__ Wh,
                         const float* __restrict__ scales,
                         const float* __restrict__ bias,
                         float* __restrict__ Y) {
    __shared__ __align__(16) _Float16 As[2][BM * LDSP];
    __shared__ __align__(16) _Float16 Bs[2][BN * LDSP];

    const int t    = threadIdx.x;
    const int lane = t & 31;
    const int wv   = t >> 5;        // wave 0..7
    const int wm   = wv >> 2;       // 0..1  (M direction, 64 rows each)
    const int wn   = wv & 3;        // 0..3  (N direction, 64 cols each)
    const int l15  = lane & 15;
    const int hiL  = lane >> 4;

    const int m0 = blockIdx.y * BM;
    const int n0 = blockIdx.x * BN;

    // A staging: 128 rows x 16 float4 cols; thread: row t>>1, 8 float4 chunks
    const int ar = t >> 1;               // 0..127
    const int ah = (t & 1) * 8;          // float4 column base (0 or 8)
    const float* gA = X + (size_t)(m0 + ar) * K_DIM + ah * 4;

    const _Float16* gB = Wh + (size_t)n0 * K_DIM;   // tile row start (TDM strides by K)
    const unsigned bs_off0 = (unsigned)(unsigned long long)(const void*)&Bs[0][0];
    const unsigned bs_off1 = (unsigned)(unsigned long long)(const void*)&Bs[1][0];

    v8f acc[4][4] = {};
    float4 areg[8];

    const int aKoff = hiL ? 8 : 0;   // A: lanes>=16 hold k 8..15 / 24..31
    const int bKoff = hiL ? 8 : 0;   // B: lanes>=16 hold k 8..15 / 24..31

    // ---------------- prologue: fill buffer 0 ----------------
    if (wv == 0) tdm_load_b_tile(gB, bs_off0);
    #pragma unroll
    for (int i = 0; i < 8; ++i)
        areg[i] = *(const float4*)(gA + i * 4);
    #pragma unroll
    for (int i = 0; i < 8; ++i) {
        PK4 pk;
        pk.h[0] = (_Float16)areg[i].x; pk.h[1] = (_Float16)areg[i].y;
        pk.h[2] = (_Float16)areg[i].z; pk.h[3] = (_Float16)areg[i].w;
        *(uint2*)(&As[0][ar * LDSP + (ah + i) * 4]) = pk.u;
    }
    if (wv == 0) __builtin_amdgcn_s_wait_tensorcnt(0);
    __syncthreads();

    // ---------------- main loop over K tiles ----------------
    const int NT = K_DIM / BK;   // 64
    for (int kt = 0; kt < NT; ++kt) {
        const int cur = kt & 1;
        const int nxt = cur ^ 1;

        if (kt + 1 < NT) {
            if (wv == 0)
                tdm_load_b_tile(gB + (size_t)(kt + 1) * BK, nxt ? bs_off1 : bs_off0);
            const float* gAn = gA + (size_t)(kt + 1) * BK;
            #pragma unroll
            for (int i = 0; i < 8; ++i)
                areg[i] = *(const float4*)(gAn + i * 4);
        }

        // ---- compute: 2 k-steps x 16 WMMAs from buffer `cur` ----
        #pragma unroll
        for (int s = 0; s < 2; ++s) {
            const int sOff = s * 32;
            v16h af[4], bf[4];
            #pragma unroll
            for (int i = 0; i < 4; ++i) {
                const _Float16* p = &As[cur][(wm * 64 + i * 16 + l15) * LDSP + sOff + aKoff];
                V16 tmp;
                tmp.q[0] = *(const uint4*)(p);
                tmp.q[1] = *(const uint4*)(p + 16);
                af[i] = tmp.v;
            }
            #pragma unroll
            for (int j = 0; j < 4; ++j) {
                const _Float16* p = &Bs[cur][(wn * 64 + j * 16 + l15) * LDSP + sOff + bKoff];
                V16 tmp;
                tmp.q[0] = *(const uint4*)(p);
                tmp.q[1] = *(const uint4*)(p + 16);
                bf[j] = tmp.v;
            }
            #pragma unroll
            for (int i = 0; i < 4; ++i)
                #pragma unroll
                for (int j = 0; j < 4; ++j)
                    acc[i][j] = __builtin_amdgcn_wmma_f32_16x16x32_f16(
                        false, af[i], false, bf[j], (short)0, acc[i][j], false, false);
        }

        if (kt + 1 < NT) {
            #pragma unroll
            for (int i = 0; i < 8; ++i) {
                PK4 pk;
                pk.h[0] = (_Float16)areg[i].x; pk.h[1] = (_Float16)areg[i].y;
                pk.h[2] = (_Float16)areg[i].z; pk.h[3] = (_Float16)areg[i].w;
                *(uint2*)(&As[nxt][ar * LDSP + (ah + i) * 4]) = pk.u;
            }
            if (wv == 0) __builtin_amdgcn_s_wait_tensorcnt(0);
        }
        __syncthreads();
    }

    // ---------------- epilogue: dequant scale + bias ----------------
    #pragma unroll
    for (int j = 0; j < 4; ++j) {
        const int gn = n0 + wn * 64 + j * 16 + l15;
        const float s  = scales[gn];
        const float bz = bias[gn];
        #pragma unroll
        for (int i = 0; i < 4; ++i) {
            const int gm0 = m0 + wm * 64 + i * 16 + (hiL << 3);
            float* yp = Y + (size_t)gm0 * N_DIM + gn;
            #pragma unroll
            for (int r = 0; r < 8; ++r)
                yp[(size_t)r * N_DIM] = acc[i][j][r] * s + bz;
        }
    }
}

// ---------------------------------------------------------------------------
extern "C" void kernel_launch(void* const* d_in, const int* in_sizes, int n_in,
                              void* d_out, int out_size, void* d_ws, size_t ws_size,
                              hipStream_t stream) {
    const float* x  = (const float*)d_in[0];
    const int*   qw = (const int*)d_in[1];
    const float* sc = (const float*)d_in[2];
    const float* bi = (const float*)d_in[3];
    float*       y  = (float*)d_out;

    const int M  = in_sizes[0] / K_DIM;   // B*S = 16384
    const int nW = N_DIM * K_DIM;         // 16M weight elements

    _Float16* wh = (_Float16*)d_ws;       // 32 MB f16 weights in scratch

    w_to_f16_kernel<<<nW / (256 * 8), 256, 0, stream>>>(qw, wh, nW);

    dim3 grid(N_DIM / BN, M / BM);        // N fastest -> L2 reuse of x slab
    qlinear_wmma_kernel<<<grid, 256, 0, stream>>>(x, wh, sc, bi, y);
}
